// MultiHeadAttention_78726750535799
// MI455X (gfx1250) — matrix operand order repaired and checked by reading, not compiled, not run on hardware
//
#include <hip/hip_runtime.h>
#include <hip/hip_bf16.h>

typedef __bf16 bf16_t;
typedef __attribute__((ext_vector_type(16))) __bf16 v16bf;
typedef __attribute__((ext_vector_type(8)))  float  v8f;
typedef unsigned int u32x4 __attribute__((ext_vector_type(4)));
typedef int          i32x8 __attribute__((ext_vector_type(8)));
typedef int          i32x4 __attribute__((ext_vector_type(4)));

#define NUM_HEADS 16
#define HEAD_DIM  64
#define INNER     1024   // NUM_HEADS * HEAD_DIM
#define LQ        4096
#define LK        4096
#define TAU       0.125f // 64^-0.5

#if defined(__has_builtin)
#  if __has_builtin(__builtin_amdgcn_tensor_load_to_lds) && __has_builtin(__builtin_amdgcn_s_wait_tensorcnt)
#    define HAVE_TDM 1
#  endif
#endif
#ifndef HAVE_TDM
#  define HAVE_TDM 0
#endif

union ABFrag {
    v16bf v;
    uint4 q[2];
};
union BfPair {
    bf16_t h[2];
    unsigned u;
};

// ---- fragment loaders from LDS (row-major tiles, 16B-aligned rows) ----
// A 16x32 bf16 (ISA 7.12.2): lane m=l16; VGPR0-3: K = 8*half + {0..7};
// VGPR4-7: K = 16 + 8*half + {0..7}
__device__ __forceinline__ v16bf load_fragA(const bf16_t* row, int half) {
    ABFrag f;
    f.q[0] = *(const uint4*)(row + 8 * half);
    f.q[1] = *(const uint4*)(row + 16 + 8 * half);
    return f.v;
}
// B 32x16 bf16: lane n=l16; VGPR0-3: K = 8*half + {0..7}; VGPR4-7: K = 16 + 8*half + {0..7}
__device__ __forceinline__ v16bf load_fragB(const bf16_t* row, int half) {
    ABFrag f;
    f.q[0] = *(const uint4*)(row + 8 * half);
    f.q[1] = *(const uint4*)(row + 16 + 8 * half);
    return f.v;
}

__device__ __forceinline__ v8f wmma_bf16(v16bf a, v16bf b, v8f c) {
    return __builtin_amdgcn_wmma_f32_16x16x32_bf16(
        false, a, false, b, (short)0, c, false, false);
}

__device__ __forceinline__ float rmax16(float v) {
#pragma unroll
    for (int m = 1; m < 16; m <<= 1) v = fmaxf(v, __shfl_xor(v, m, 32));
    return v;
}
__device__ __forceinline__ float rsum16(float v) {
#pragma unroll
    for (int m = 1; m < 16; m <<= 1) v += __shfl_xor(v, m, 32);
    return v;
}

#if HAVE_TDM
// TDM: DMA one 64x64 bf16 tile (row-major, row stride in elements) into LDS.
// D# packing per CDNA5 ISA:
//   group0: count=1 | lds_addr | global_addr[56:0] | type=2
//   group1: data_size=2B, tensor_dim0/1 = tile_dim0/1 = 64, dim0 stride.
__device__ __forceinline__ void tdm_load_tile_64x64_bf16(
    unsigned lds_byte_addr, const bf16_t* gptr, unsigned row_stride_elems)
{
    unsigned long long ga = (unsigned long long)(uintptr_t)gptr;
    u32x4 g0;
    g0.x = 1u;                                   // count=1, user mode
    g0.y = lds_byte_addr;                        // LDS dest (bytes)
    g0.z = (unsigned)ga;                         // global_addr[31:0]
    g0.w = (unsigned)(ga >> 32) | 0x80000000u;   // global_addr[56:32] | type=2
    i32x8 g1;
    g1[0] = 0x00010000;                          // data_size = 2 bytes
    g1[1] = (64 << 16);                          // tensor_dim0 = 64 (low16)
    g1[2] = (64 << 16);                          // tensor_dim0 hi | tensor_dim1 = 64
    g1[3] = (64 << 16);                          // tensor_dim1 hi | tile_dim0 = 64
    g1[4] = 64;                                  // tile_dim1 = 64, tile_dim2 = 0
    g1[5] = (int)row_stride_elems;               // tensor_dim0_stride low32
    g1[6] = 0;
    g1[7] = 0;
    i32x4 z4 = {0, 0, 0, 0};
#if defined(__clang_major__) && __clang_major__ >= 23
    i32x8 z8 = {0, 0, 0, 0, 0, 0, 0, 0};
    __builtin_amdgcn_tensor_load_to_lds(g0, g1, z4, z4, z8, 0);
#else
    __builtin_amdgcn_tensor_load_to_lds(g0, g1, z4, z4, 0);
#endif
}
__device__ __forceinline__ unsigned lds_off(const void* p) {
    return (unsigned)(uintptr_t)p;   // low 32 bits of LDS-aperture address = LDS offset
}
#endif

// copy one 64x64 bf16 tile (contiguous rows) global -> LDS, 128 threads
__device__ __forceinline__ void stage_tile_64x64(
    bf16_t (*dst)[64], const bf16_t* src, size_t row_stride, int tid)
{
    const int r = tid >> 1, cs = (tid & 1) * 32;
    const uint4* s = (const uint4*)(src + (size_t)r * row_stride + cs);
    uint4* d = (uint4*)&dst[r][cs];
    d[0] = s[0]; d[1] = s[1]; d[2] = s[2]; d[3] = s[3];
}

// =====================================================================
// GEMM: C[M,N] = (A[M,K] @ B[K,N] + bias[N]) * scale
// =====================================================================
template <bool A_BF16, bool OUT_BF16>
__global__ void __launch_bounds__(128) gemm_bias_kernel(
    const void* __restrict__ Aptr, const float* __restrict__ B,
    const float* __restrict__ bias, void* __restrict__ Cptr,
    int M, int N, int K, float scale)
{
    __shared__ __align__(16) bf16_t Asl[64][32];
    __shared__ __align__(16) bf16_t BslT[64][32]; // [n][k]

    const int tid  = threadIdx.x;
    const int w    = tid >> 5;
    const int lane = tid & 31;
    const int half = lane >> 4;
    const int l16  = lane & 15;
    const int wr   = w >> 1, wc = w & 1;
    const int m0 = blockIdx.x * 64, n0 = blockIdx.y * 64;

    const v8f vzero = {0.f, 0.f, 0.f, 0.f, 0.f, 0.f, 0.f, 0.f};
    v8f acc[2][2];
#pragma unroll
    for (int i = 0; i < 2; ++i)
#pragma unroll
        for (int j = 0; j < 2; ++j) acc[i][j] = vzero;

    for (int k0 = 0; k0 < K; k0 += 32) {
        __syncthreads();
        // ---- stage A tile 64x32 (bf16) ----
        {
            const int r = tid >> 1, cs = (tid & 1) * 16;
            if constexpr (A_BF16) {
                const bf16_t* src = (const bf16_t*)Aptr + (size_t)(m0 + r) * K + k0 + cs;
                *(uint4*)&Asl[r][cs]     = *(const uint4*)(src);
                *(uint4*)&Asl[r][cs + 8] = *(const uint4*)(src + 8);
            } else {
                const float* src = (const float*)Aptr + (size_t)(m0 + r) * K + k0 + cs;
                union { bf16_t h[16]; uint4 q[2]; } T;
#pragma unroll
                for (int i = 0; i < 16; ++i) T.h[i] = (bf16_t)src[i];
                *(uint4*)&Asl[r][cs]     = T.q[0];
                *(uint4*)&Asl[r][cs + 8] = T.q[1];
            }
        }
        // ---- stage B tile 32x64 transposed, packed b32 stores ----
        {
            const int kk = (tid >> 3) * 2, nb = (tid & 7) * 8;
            const float* s0 = B + (size_t)(k0 + kk) * N + n0 + nb;
            const float* s1 = s0 + N;
#pragma unroll
            for (int i = 0; i < 8; ++i) {
                BfPair p;
                p.h[0] = (bf16_t)s0[i];
                p.h[1] = (bf16_t)s1[i];
                *(unsigned*)&BslT[nb + i][kk] = p.u;
            }
        }
        __syncthreads();

        v16bf afrag[2], bfrag[2];
#pragma unroll
        for (int i = 0; i < 2; ++i)
            afrag[i] = load_fragA(&Asl[wr * 32 + i * 16 + l16][0], half);
#pragma unroll
        for (int j = 0; j < 2; ++j)
            bfrag[j] = load_fragB(&BslT[wc * 32 + j * 16 + l16][0], half);
#pragma unroll
        for (int i = 0; i < 2; ++i)
#pragma unroll
            for (int j = 0; j < 2; ++j)
                acc[i][j] = wmma_bf16(afrag[i], bfrag[j], acc[i][j]);
    }

#pragma unroll
    for (int i = 0; i < 2; ++i) {
#pragma unroll
        for (int j = 0; j < 2; ++j) {
            const int n = n0 + wc * 32 + j * 16 + l16;
            const float bv = bias[n];
#pragma unroll
            for (int r = 0; r < 8; ++r) {
                const int m = m0 + wr * 32 + i * 16 + r + 8 * half;
                const float val = (acc[i][j][r] + bv) * scale;
                if constexpr (OUT_BF16)
                    ((bf16_t*)Cptr)[(size_t)m * N + n] = (bf16_t)val;
                else
                    ((float*)Cptr)[(size_t)m * N + n] = val;
            }
        }
    }
}

// =====================================================================
// One-time V transpose: VhT[h][d][t] = Vh[t][h*64+d]   (bf16, 16 MB traffic)
// =====================================================================
__global__ void __launch_bounds__(128) transpose_v_kernel(
    const bf16_t* __restrict__ Vh, bf16_t* __restrict__ VhT)
{
    __shared__ __align__(16) bf16_t T[64][64];
    const int h = blockIdx.y, t0 = blockIdx.x * 64;
    const int tid = threadIdx.x;
    stage_tile_64x64(T, Vh + (size_t)t0 * INNER + h * HEAD_DIM, INNER, tid);
    __syncthreads();
    {
        const int d = tid >> 1, ts = (tid & 1) * 32;
        unsigned buf[16] __attribute__((aligned(16)));
#pragma unroll
        for (int i = 0; i < 16; ++i) {
            BfPair p;
            p.h[0] = T[ts + 2 * i][d];
            p.h[1] = T[ts + 2 * i + 1][d];
            buf[i] = p.u;
        }
        uint4* dst = (uint4*)(VhT + ((size_t)h * HEAD_DIM + d) * LK + t0 + ts);
#pragma unroll
        for (int i = 0; i < 4; ++i) dst[i] = ((const uint4*)buf)[i];
    }
}

// =====================================================================
// Attention: per (head, 64-row q tile). Two-pass recompute softmax.
// Pass 1: per-lane online max/sum (no cross-lane traffic in the loop),
//         single 16-lane combine at the end.
// Pass 2: recompute S, write normalized probs to att (1 GB, exactly once),
//         accumulate ctx = P @ V via WMMA. Tiles staged via TDM.
// =====================================================================
__global__ void __launch_bounds__(128) attn_kernel(
    const bf16_t* __restrict__ Qh, const bf16_t* __restrict__ Kh,
    const bf16_t* __restrict__ VhT, float* __restrict__ att,
    bf16_t* __restrict__ ctx)
{
    const int h  = blockIdx.y;
    const int q0 = blockIdx.x * 64;
    const int tid  = threadIdx.x;
    const int w    = tid >> 5;
    const int lane = tid & 31;
    const int half = lane >> 4;
    const int l16  = lane & 15;

    __shared__ __align__(16) bf16_t Qt[64][64];
    __shared__ __align__(16) bf16_t Kt[64][64];   // [key][dim]  == B^T for QK^T
    __shared__ __align__(16) bf16_t VtT[64][64];  // [dim][key]  == B^T for P@V
    __shared__ __align__(16) bf16_t Pt[64][64];   // probs, A layout for P@V

    const v8f vzero = {0.f, 0.f, 0.f, 0.f, 0.f, 0.f, 0.f, 0.f};
    const bf16_t* Qg  = Qh + (size_t)q0 * INNER + h * HEAD_DIM;
    const bf16_t* Vtg = VhT + (size_t)h * HEAD_DIM * LK;   // rows: dim, cols: key

    // ---- stage Q tile ----
#if HAVE_TDM
    if (tid < 32) {
        tdm_load_tile_64x64_bf16(lds_off(&Qt[0][0]), Qg, INNER);
        __builtin_amdgcn_s_wait_tensorcnt(0);
    }
#else
    stage_tile_64x64(Qt, Qg, INNER, tid);
#endif
    __syncthreads();

    v16bf aq[2];
#pragma unroll
    for (int s = 0; s < 2; ++s)
        aq[s] = load_fragA(&Qt[w * 16 + l16][s * 32], half);

    // per-lane online softmax state (rows r+8*half of this wave's 16 rows)
    float m_ln[8], l_ln[8];
#pragma unroll
    for (int r = 0; r < 8; ++r) { m_ln[r] = -__builtin_inff(); l_ln[r] = 0.f; }

    // =================== PASS 1 ===================
    for (int kt = 0; kt < LK; kt += 64) {
        __syncthreads();
#if HAVE_TDM
        if (tid < 32) {
            tdm_load_tile_64x64_bf16(lds_off(&Kt[0][0]),
                                     Kh + (size_t)kt * INNER + h * HEAD_DIM, INNER);
            __builtin_amdgcn_s_wait_tensorcnt(0);
        }
#else
        stage_tile_64x64(Kt, Kh + (size_t)kt * INNER + h * HEAD_DIM, INNER, tid);
#endif
        __syncthreads();

#pragma unroll
        for (int c = 0; c < 4; ++c) {
            v8f s_acc = vzero;
#pragma unroll
            for (int s = 0; s < 2; ++s) {
                v16bf bk = load_fragB(&Kt[c * 16 + l16][s * 32], half);
                s_acc = wmma_bf16(aq[s], bk, s_acc);
            }
#pragma unroll
            for (int r = 0; r < 8; ++r) {   // per-lane online update, no shuffles
                const float sv = s_acc[r];
                const float mn = fmaxf(m_ln[r], sv);
                l_ln[r] = l_ln[r] * __expf(m_ln[r] - mn) + __expf(sv - mn);
                m_ln[r] = mn;
            }
        }
    }

    // single cross-lane combine (16 cols per half)
    float mrow[8], linv[8];
#pragma unroll
    for (int r = 0; r < 8; ++r) {
        const float M = rmax16(m_ln[r]);
        const float L = rsum16(l_ln[r] * __expf(m_ln[r] - M));
        mrow[r] = M;
        linv[r] = 1.0f / L;
    }

    v8f ctxacc[4];
#pragma unroll
    for (int c = 0; c < 4; ++c) ctxacc[c] = vzero;

    // =================== PASS 2 ===================
    for (int kt = 0; kt < LK; kt += 64) {
        __syncthreads();
#if HAVE_TDM
        if (tid < 32) {
            tdm_load_tile_64x64_bf16(lds_off(&Kt[0][0]),
                                     Kh + (size_t)kt * INNER + h * HEAD_DIM, INNER);
            tdm_load_tile_64x64_bf16(lds_off(&VtT[0][0]), Vtg + kt, LK);
            __builtin_amdgcn_s_wait_tensorcnt(0);
        }
#else
        stage_tile_64x64(Kt, Kh + (size_t)kt * INNER + h * HEAD_DIM, INNER, tid);
        stage_tile_64x64(VtT, Vtg + kt, LK, tid);
#endif
        __syncthreads();

#pragma unroll
        for (int c = 0; c < 4; ++c) {
            v8f s_acc = vzero;
#pragma unroll
            for (int s = 0; s < 2; ++s) {
                v16bf bk = load_fragB(&Kt[c * 16 + l16][s * 32], half);
                s_acc = wmma_bf16(aq[s], bk, s_acc);
            }
#pragma unroll
            for (int r = 0; r < 8; ++r) {
                const float p  = __expf(s_acc[r] - mrow[r]) * linv[r];
                const int   gm = q0 + w * 16 + r + 8 * half;
                const int   gn = kt + c * 16 + l16;
                att[((size_t)h * LQ + gm) * LK + gn] = p;   // 1 GB output, once
                Pt[w * 16 + r + 8 * half][c * 16 + l16] = (bf16_t)p;
            }
        }
        // ctx += P @ V  (wave reads only its own Pt rows; LDS in-order per wave)
        v16bf ap[2];
#pragma unroll
        for (int s = 0; s < 2; ++s)
            ap[s] = load_fragA(&Pt[w * 16 + l16][s * 32], half);
#pragma unroll
        for (int c = 0; c < 4; ++c) {
#pragma unroll
            for (int s = 0; s < 2; ++s) {
                v16bf bv = load_fragB(&VtT[c * 16 + l16][s * 32], half);
                ctxacc[c] = wmma_bf16(ap[s], bv, ctxacc[c]);
            }
        }
    }

#pragma unroll
    for (int c = 0; c < 4; ++c) {
#pragma unroll
        for (int r = 0; r < 8; ++r) {
            const int gm = q0 + w * 16 + r + 8 * half;
            ctx[(size_t)gm * INNER + h * HEAD_DIM + c * 16 + l16] = (bf16_t)ctxacc[c][r];
        }
    }
}

// =====================================================================
extern "C" void kernel_launch(void* const* d_in, const int* in_sizes, int n_in,
                              void* d_out, int out_size, void* d_ws, size_t ws_size,
                              hipStream_t stream) {
    const float* q  = (const float*)d_in[0];
    const float* k  = (const float*)d_in[1];
    const float* v  = (const float*)d_in[2];
    const float* Wq = (const float*)d_in[3];
    const float* bq = (const float*)d_in[4];
    const float* Wk = (const float*)d_in[5];
    const float* bk = (const float*)d_in[6];
    const float* Wv = (const float*)d_in[7];
    const float* bv = (const float*)d_in[8];
    const float* Wo = (const float*)d_in[9];
    const float* bo = (const float*)d_in[10];

    float* out = (float*)d_out;                       // [LQ, 1024]
    float* att = out + (size_t)LQ * INNER;            // [h, LQ, LK]

    const size_t proj = (size_t)LQ * INNER;           // elements
    bf16_t* Qh  = (bf16_t*)d_ws;
    bf16_t* Kh  = Qh + proj;
    bf16_t* Vh  = Kh + proj;
    bf16_t* VhT = Vh + proj;
    bf16_t* ctx = VhT + proj;                         // 40 MB total in d_ws

    dim3 gblk(LQ / 64, INNER / 64);                   // 64 x 16
    gemm_bias_kernel<false, true><<<gblk, 128, 0, stream>>>(q, Wq, bq, Qh, LQ, INNER, 1024, TAU);
    gemm_bias_kernel<false, true><<<gblk, 128, 0, stream>>>(k, Wk, bk, Kh, LK, INNER, 1024, 1.0f);
    gemm_bias_kernel<false, true><<<gblk, 128, 0, stream>>>(v, Wv, bv, Vh, LK, INNER, 1024, 1.0f);

    transpose_v_kernel<<<dim3(LK / 64, NUM_HEADS), 128, 0, stream>>>(Vh, VhT);

    attn_kernel<<<dim3(LQ / 64, NUM_HEADS), 128, 0, stream>>>(Qh, Kh, VhT, att, ctx);

    gemm_bias_kernel<true, false><<<gblk, 128, 0, stream>>>(ctx, Wo, bo, out, LQ, 1024, INNER, 1.0f);
}
